// DialogueRNNCell_39384850104864
// MI455X (gfx1250) — matrix operand order repaired and checked by reading, not compiled, not run on hardware
//
#include <hip/hip_runtime.h>
#include <hip/hip_bf16.h>

// ---------------------------------------------------------------------------
// DialogueRNN cell for MI455X (gfx1250, wave32).
// - 6 GEMMs on v_wmma_f32_16x16x32_bf16 (bf16 weights converted once, L2-resident)
// - attention stages g_hist[:,b,:] (256 KB) into LDS via gfx1250 async copies
//   (global_load_async_to_lds_b128 / s_wait_asynccnt) so g_hist is read from
//   DRAM exactly once instead of twice.
// - unconditional speculative global_prefetch_b8 of the next K tile (ISA: bad
//   translations are silently dropped, so no guard/branches needed).
// ---------------------------------------------------------------------------

typedef __attribute__((ext_vector_type(16))) __bf16 v16bf;
typedef __attribute__((ext_vector_type(8)))  __bf16 v8bf;
typedef __attribute__((ext_vector_type(8)))  float  v8f;

#define DIM   1024   // D_M = D_G = D_P = D_E
#define NB    1024   // batch
#define TT    64     // time steps
#define N3    3072   // 3 * hidden

// ---------------- f32 -> bf16 conversion ----------------
__global__ void k_cvt_bf16(const float* __restrict__ src, __bf16* __restrict__ dst, int n) {
  int i = blockIdx.x * blockDim.x + threadIdx.x;
  if (i < n) dst[i] = (__bf16)src[i];
}

// x_g = [U | q0[b, argmax(qmask[b])]] as bf16, shape (B, 2048)
__global__ void k_build_xg(const float* __restrict__ U, const float* __restrict__ qmask,
                           const float* __restrict__ q0, __bf16* __restrict__ xg) {
  int i = blockIdx.x * blockDim.x + threadIdx.x;   // B*2048
  int b = i >> 11, j = i & 2047;
  int sel = qmask[b * 2 + 1] > qmask[b * 2 + 0] ? 1 : 0;
  float v = (j < DIM) ? U[(size_t)b * DIM + j]
                      : q0[((size_t)b * 2 + sel) * DIM + (j - DIM)];
  xg[i] = (__bf16)v;
}

// x_p rows (b*2+p) = [U[b] | c[b]] as bf16, shape (2B, 2048)
__global__ void k_build_xp(const float* __restrict__ U, const float* __restrict__ c,
                           __bf16* __restrict__ xp) {
  int i = blockIdx.x * blockDim.x + threadIdx.x;   // 2B*2048
  int row = i >> 11, j = i & 2047;
  int b = row >> 1;
  float v = (j < DIM) ? U[(size_t)b * DIM + j] : c[(size_t)b * DIM + (j - DIM)];
  xp[i] = (__bf16)v;
}

// ---------------- WMMA bf16 GEMM:  C(MxN) = A(MxK) * W(NxK)^T + bias ----------------
// Block: 256 threads = 8 waves; block tile 64(M) x 256(N); wave tile 32x64
// (2 A-frags x 4 B-frags -> 8 v_wmma per 32-wide K step).
__global__ __launch_bounds__(256) void
k_gemm_bf16(const __bf16* __restrict__ A,   // M x K row-major
            const __bf16* __restrict__ W,   // N x K row-major (B-matrix = W^T)
            const float*  __restrict__ bias,// length N
            float* __restrict__ C,          // M x N row-major
            int M, int N, int K)
{
  const int lane = threadIdx.x & 31;
  const int wave = threadIdx.x >> 5;   // 0..7
  const int wm   = wave >> 2;          // 0..1
  const int wn   = wave & 3;           // 0..3
  const int m0   = blockIdx.y * 64  + wm * 32;
  const int n0   = blockIdx.x * 256 + wn * 64;
  const int l15  = lane & 15;
  const int half = lane >> 4;

  const v8f vzero = {0.f, 0.f, 0.f, 0.f, 0.f, 0.f, 0.f, 0.f};
  v8f acc[2][4];
#pragma unroll
  for (int mi = 0; mi < 2; ++mi)
#pragma unroll
    for (int ni = 0; ni < 4; ++ni) acc[mi][ni] = vzero;

  for (int k0 = 0; k0 < K; k0 += 32) {
    // A fragments: 16x32 bf16.  Lane (half,m) -> K = k0 + half*16 + i (contiguous)
    v16bf afr[2];
#pragma unroll
    for (int mi = 0; mi < 2; ++mi) {
      const __bf16* ap = A + (size_t)(m0 + mi * 16 + l15) * K + k0 + half * 16;
      __builtin_prefetch(ap + 32, 0, 3);   // speculative global_prefetch_b8, no guard
      afr[mi] = *(const v16bf*)ap;         // 32B aligned
    }
    // B fragments: 32x16 bf16.  Lane n = l15, K = k0 + half*16 + i  (contiguous in W row)
    v16bf bfr[4];
#pragma unroll
    for (int ni = 0; ni < 4; ++ni) {
      const __bf16* wp = W + (size_t)(n0 + ni * 16 + l15) * K + k0 + half * 16;
      if (ni == 0) __builtin_prefetch(wp + 32, 0, 3);
      bfr[ni] = *(const v16bf*)wp;         // 32B aligned
    }
#pragma unroll
    for (int mi = 0; mi < 2; ++mi)
#pragma unroll
      for (int ni = 0; ni < 4; ++ni)
        acc[mi][ni] = __builtin_amdgcn_wmma_f32_16x16x32_bf16(
            false, afr[mi], false, bfr[ni], (short)0, acc[mi][ni], false, false);
  }

  // Epilogue: D layout -> row = m0 + mi*16 + half*8 + r, col = n0 + ni*16 + l15
#pragma unroll
  for (int mi = 0; mi < 2; ++mi)
#pragma unroll
    for (int ni = 0; ni < 4; ++ni) {
      const int col = n0 + ni * 16 + l15;
      const float bv = bias[col];
#pragma unroll
      for (int r = 0; r < 8; ++r) {
        const int row = m0 + mi * 16 + half * 8 + r;
        C[(size_t)row * N + col] = acc[mi][ni][r] + bv;
      }
    }
}

// ---------------- GRU gate combine ----------------
__device__ __forceinline__ float sigm(float x) { return 1.f / (1.f + __expf(-x)); }

__global__ void k_gru_combine(const float* __restrict__ gi, const float* __restrict__ gh,
                              const float* __restrict__ h, float* __restrict__ out, int rows) {
  int i = blockIdx.x * blockDim.x + threadIdx.x;   // rows * 1024
  int row = i >> 10, j = i & 1023;
  if (row >= rows) return;
  const float* gir = gi + (size_t)row * N3;
  const float* ghr = gh + (size_t)row * N3;
  float r = sigm(gir[j] + ghr[j]);
  float z = sigm(gir[DIM + j] + ghr[DIM + j]);
  float n = tanhf(gir[2 * DIM + j] + r * ghr[2 * DIM + j]);
  out[i] = (1.f - z) * n + z * h[i];
}

// p-GRU combine + party masking + bf16 gather of e-GRU input (selected party row)
__global__ void k_gru_combine_p(const float* __restrict__ gi, const float* __restrict__ gh,
                                const float* __restrict__ q0, const float* __restrict__ qmask,
                                float* __restrict__ out_q, __bf16* __restrict__ xe) {
  int i = blockIdx.x * blockDim.x + threadIdx.x;   // 2B * 1024
  int row = i >> 10, j = i & 1023;                 // row = b*2 + p
  const float* gir = gi + (size_t)row * N3;
  const float* ghr = gh + (size_t)row * N3;
  float r = sigm(gir[j] + ghr[j]);
  float z = sigm(gir[DIM + j] + ghr[DIM + j]);
  float n = tanhf(gir[2 * DIM + j] + r * ghr[2 * DIM + j]);
  float qs = (1.f - z) * n + z * q0[i];
  float m = qmask[row];                            // one-hot over party
  float qv = q0[i] * (1.f - m) + qs * m;
  out_q[i] = qv;
  if (m > 0.5f) xe[((size_t)(row >> 1) << 10) + j] = (__bf16)qv;   // q_[b, qm_idx[b]]
}

// ---------------- attention: one workgroup per batch row ----------------
// g_hist[:,b,:] (64 x 1024 f32 = 256 KB) is staged into LDS with gfx1250
// async global->LDS copies, then both passes (scale dot + weighted context)
// read LDS.  512 threads = 16 waves; 1 workgroup per WGP (256 KB LDS).
__global__ __launch_bounds__(512) void
k_attention(const float* __restrict__ ghist, const float* __restrict__ attn_w,
            float* __restrict__ cvec, float* __restrict__ alpha_out) {
  const int b = blockIdx.x;
  const int tid = threadIdx.x;
  const int lane = tid & 31, wave = tid >> 5;      // 16 waves
  extern __shared__ float sg[];                    // TT*DIM floats (256 KB)
  __shared__ float s_scale[TT];
  __shared__ float s_alpha[TT];

  // ---- stage: async copy g_hist[:,b,:] -> LDS (tracked by ASYNCcnt) ----
  // 64 rows x 256 float4 = 16384 b128 transfers; 32 per thread, no divergence.
  for (int idx = tid; idx < TT * (DIM / 4); idx += 512) {
    const int t = idx >> 8;                        // row (timestep)
    const int c = idx & 255;                       // float4 within row
    // low 32 bits of a flat LDS pointer == LDS byte offset (ISA 10.2)
    unsigned lds_off = (unsigned)(size_t)&sg[(t << 10) + (c << 2)];
    const float* gp = ghist + ((size_t)t * NB + b) * DIM + (c << 2);
    asm volatile("global_load_async_to_lds_b128 %0, %1, off"
                 :: "v"(lds_off), "v"(gp) : "memory");
  }
  asm volatile("s_wait_asynccnt 0x0" ::: "memory");
  __syncthreads();

  // ---- phase 1: scale[t] = dot(sg[t,:], attn_w)   (wave w owns 4 timesteps) ----
  for (int k = 0; k < 4; ++k) {
    int t = wave * 4 + k;
    float s = 0.f;
    for (int d = lane; d < DIM; d += 32) s += sg[(t << 10) + d] * attn_w[d];
#pragma unroll
    for (int off = 16; off > 0; off >>= 1) s += __shfl_xor(s, off, 32);
    if (lane == 0) s_scale[t] = s;
  }
  __syncthreads();

  // ---- phase 2: softmax over T=64 (first wave, 2 values/lane) ----
  if (wave == 0) {
    float v0 = s_scale[lane], v1 = s_scale[lane + 32];
    float m = fmaxf(v0, v1);
#pragma unroll
    for (int off = 16; off > 0; off >>= 1) m = fmaxf(m, __shfl_xor(m, off, 32));
    float e0 = __expf(v0 - m), e1 = __expf(v1 - m);
    float s = e0 + e1;
#pragma unroll
    for (int off = 16; off > 0; off >>= 1) s += __shfl_xor(s, off, 32);
    float inv = 1.f / s;
    s_alpha[lane]      = e0 * inv;
    s_alpha[lane + 32] = e1 * inv;
    alpha_out[(size_t)b * TT + lane]      = e0 * inv;   // alpha_out[b,0,t]
    alpha_out[(size_t)b * TT + lane + 32] = e1 * inv;
  }
  __syncthreads();

  // ---- phase 3: c[b,d] = sum_t alpha[t] * sg[t,d]  (LDS only, no 2nd DRAM pass) ----
  for (int d = tid; d < DIM; d += 512) {
    float a = 0.f;
#pragma unroll 8
    for (int t = 0; t < TT; ++t) a += s_alpha[t] * sg[(t << 10) + d];
    cvec[(size_t)b * DIM + d] = a;
  }
}

// ---------------------------------------------------------------------------
extern "C" void kernel_launch(void* const* d_in, const int* in_sizes, int n_in,
                              void* d_out, int out_size, void* d_ws, size_t ws_size,
                              hipStream_t stream) {
  // inputs, setup_inputs() order (param dicts flatten in insertion order):
  const float* U      = (const float*)d_in[0];
  const float* qmask  = (const float*)d_in[1];
  const float* ghist  = (const float*)d_in[2];
  const float* q0     = (const float*)d_in[3];
  const float* e0     = (const float*)d_in[4];
  const float* gWih   = (const float*)d_in[5];
  const float* gWhh   = (const float*)d_in[6];
  const float* gbih   = (const float*)d_in[7];
  const float* gbhh   = (const float*)d_in[8];
  const float* pWih   = (const float*)d_in[9];
  const float* pWhh   = (const float*)d_in[10];
  const float* pbih   = (const float*)d_in[11];
  const float* pbhh   = (const float*)d_in[12];
  const float* eWih   = (const float*)d_in[13];
  const float* eWhh   = (const float*)d_in[14];
  const float* ebih   = (const float*)d_in[15];
  const float* ebhh   = (const float*)d_in[16];
  const float* attn_w = (const float*)d_in[17];

  float* out     = (float*)d_out;
  float* out_g   = out;                              // (B, 1024)
  float* out_q   = out + (size_t)NB * DIM;           // (B, 2, 1024)
  float* out_e   = out + (size_t)3 * NB * DIM;       // (B, 1024)
  float* out_al  = out + (size_t)4 * NB * DIM;       // (B, 1, 64)

  // ---- workspace carve-up (all sizes multiples of 256B) ----
  char* ws = (char*)d_ws;
  auto take = [&](size_t bytes) { char* p = ws; ws += bytes; return p; };
  __bf16* WihG = (__bf16*)take((size_t)N3 * 2048 * 2);
  __bf16* WhhG = (__bf16*)take((size_t)N3 * DIM * 2);
  __bf16* WihP = (__bf16*)take((size_t)N3 * 2048 * 2);
  __bf16* WhhP = (__bf16*)take((size_t)N3 * DIM * 2);
  __bf16* WihE = (__bf16*)take((size_t)N3 * DIM * 2);
  __bf16* WhhE = (__bf16*)take((size_t)N3 * DIM * 2);
  __bf16* xg   = (__bf16*)take((size_t)NB * 2048 * 2);
  __bf16* hg   = (__bf16*)take((size_t)NB * DIM * 2);
  __bf16* q0b  = (__bf16*)take((size_t)2 * NB * DIM * 2);
  __bf16* e0b  = (__bf16*)take((size_t)NB * DIM * 2);
  __bf16* xp   = (__bf16*)take((size_t)2 * NB * 2048 * 2);
  __bf16* xe   = (__bf16*)take((size_t)NB * DIM * 2);
  float*  cvec = (float*) take((size_t)NB * DIM * 4);
  float*  GI   = (float*) take((size_t)2 * NB * N3 * 4);
  float*  GH   = (float*) take((size_t)2 * NB * N3 * 4);

  const float* ghist_last = ghist + (size_t)(TT - 1) * NB * DIM;

  auto cvt = [&](const float* s, __bf16* d, size_t n) {
    k_cvt_bf16<<<dim3((unsigned)((n + 255) / 256)), dim3(256), 0, stream>>>(s, d, (int)n);
  };

  // ---- stage 0: one-shot bf16 conversions (weights become L2-resident) ----
  cvt(gWih, WihG, (size_t)N3 * 2048);
  cvt(gWhh, WhhG, (size_t)N3 * DIM);
  cvt(pWih, WihP, (size_t)N3 * 2048);
  cvt(pWhh, WhhP, (size_t)N3 * DIM);
  cvt(eWih, WihE, (size_t)N3 * DIM);
  cvt(eWhh, WhhE, (size_t)N3 * DIM);
  cvt(ghist_last, hg, (size_t)NB * DIM);
  cvt(q0, q0b, (size_t)2 * NB * DIM);
  cvt(e0, e0b, (size_t)NB * DIM);

  // ---- g-GRU ----
  k_build_xg<<<dim3(NB * 2048 / 256), dim3(256), 0, stream>>>(U, qmask, q0, xg);
  k_gemm_bf16<<<dim3(N3 / 256, NB / 64), dim3(256), 0, stream>>>(xg, WihG, gbih, GI, NB, N3, 2048);
  k_gemm_bf16<<<dim3(N3 / 256, NB / 64), dim3(256), 0, stream>>>(hg, WhhG, gbhh, GH, NB, N3, DIM);
  k_gru_combine<<<dim3(NB * DIM / 256), dim3(256), 0, stream>>>(GI, GH, ghist_last, out_g, NB);

  // ---- attention (256 KB dynamic LDS per workgroup) ----
  k_attention<<<dim3(NB), dim3(512), TT * DIM * sizeof(float), stream>>>(ghist, attn_w, cvec, out_al);

  // ---- p-GRU (2B rows) ----
  k_build_xp<<<dim3(2 * NB * 2048 / 256), dim3(256), 0, stream>>>(U, cvec, xp);
  k_gemm_bf16<<<dim3(N3 / 256, 2 * NB / 64), dim3(256), 0, stream>>>(xp, WihP, pbih, GI, 2 * NB, N3, 2048);
  k_gemm_bf16<<<dim3(N3 / 256, 2 * NB / 64), dim3(256), 0, stream>>>(q0b, WhhP, pbhh, GH, 2 * NB, N3, DIM);
  k_gru_combine_p<<<dim3(2 * NB * DIM / 256), dim3(256), 0, stream>>>(GI, GH, q0, qmask, out_q, xe);

  // ---- e-GRU ----
  k_gemm_bf16<<<dim3(N3 / 256, NB / 64), dim3(256), 0, stream>>>(xe, WihE, ebih, GI, NB, N3, DIM);
  k_gemm_bf16<<<dim3(N3 / 256, NB / 64), dim3(256), 0, stream>>>(e0b, WhhE, ebhh, GH, NB, N3, DIM);
  k_gru_combine<<<dim3(NB * DIM / 256), dim3(256), 0, stream>>>(GI, GH, e0, out_e, NB);
}
